// BarrierNetDOpp_52158082842729
// MI455X (gfx1250) — matrix operand order repaired and checked by reading, not compiled, not run on hardware
//
#include <hip/hip_runtime.h>
#include <hip/hip_bf16.h>
#include <math.h>

typedef _Float16 v16h __attribute__((ext_vector_type(16)));
typedef _Float16 h8v  __attribute__((ext_vector_type(8)));
typedef float    v8f  __attribute__((ext_vector_type(8)));

namespace {
constexpr int ROWS = 128;   // batch rows per workgroup
constexpr int NTHR = 256;   // 8 waves (wave32)
constexpr int H1N  = 256;
constexpr int H2N  = 128;
constexpr int FN   = 8;
constexpr int H1S  = 264;   // h1 LDS row stride (halves), padded for bank stagger
constexpr int H2S  = 136;   // h21/h22 LDS row stride (halves)
constexpr int IPM  = 25;

// ---- LDS layout (bytes) ----
constexpr int OFF_W21F = 0;                              // 64 KB  f16 B-fragment layout
constexpr int OFF_W22F = OFF_W21F + H1N * H2N * 2;       // 64 KB
constexpr int OFF_H1   = OFF_W22F + H1N * H2N * 2;       // 128*264*2 = 66 KB
constexpr int OFF_XS   = OFF_H1 + ROWS * H1S * 2;        // 4 KB   x tile f32
constexpr int OFF_W1   = OFF_XS + ROWS * FN * 4;         // 8 KB
constexpr int OFF_B1   = OFF_W1 + FN * H1N * 4;          // 1 KB
constexpr int OFF_B21  = OFF_B1 + H1N * 4;               // 512
constexpr int OFF_B22  = OFF_B21 + H2N * 4;              // 512
constexpr int OFF_W31  = OFF_B22 + H2N * 4;              // 1 KB
constexpr int OFF_W32  = OFF_W31 + H2N * 2 * 4;          // 1 KB
constexpr int OFF_B31  = OFF_W32 + H2N * 2 * 4;          // 16
constexpr int OFF_B32  = OFF_B31 + 16;                   // 16
constexpr int OFF_MEAN = OFF_B32 + 16;                   // 32
constexpr int OFF_STD  = OFF_MEAN + 32;                  // 32
constexpr int OFF_H21  = ((OFF_STD + 32 + 15) & ~15);    // 34 KB f16
constexpr int OFF_H22  = OFF_H21 + ROWS * H2S * 2;       // 34 KB f16
constexpr int SMEMB    = OFF_H22 + ROWS * H2S * 2;       // ~278 KB < 320 KB/WGP
} // namespace

// B-fragment load: fragment(kk,nt) is 512 contiguous halves, 16 per lane.
__device__ __forceinline__ v16h load_bfrag(const _Float16* wf, int kk, int nt, int lane) {
    const _Float16* bp = wf + ((kk * 8 + nt) * 512 + lane * 16);
    h8v lo = *reinterpret_cast<const h8v*>(bp);
    h8v hi = *reinterpret_cast<const h8v*>(bp + 8);
    return __builtin_shufflevector(lo, hi, 0,1,2,3,4,5,6,7,8,9,10,11,12,13,14,15);
}

__global__ void __launch_bounds__(NTHR, 1)
barriernet_fused(const float* __restrict__ gx,
                 const float* __restrict__ gmean, const float* __restrict__ gstd,
                 const float* __restrict__ gW1,  const float* __restrict__ gb1,
                 const float* __restrict__ gW21, const float* __restrict__ gb21,
                 const float* __restrict__ gW22, const float* __restrict__ gb22,
                 const float* __restrict__ gW31, const float* __restrict__ gb31,
                 const float* __restrict__ gW32, const float* __restrict__ gb32,
                 float* __restrict__ out, int Bn)
{
    extern __shared__ char smem[];
    _Float16* w21f = reinterpret_cast<_Float16*>(smem + OFF_W21F);
    _Float16* w22f = reinterpret_cast<_Float16*>(smem + OFF_W22F);
    _Float16* h1s  = reinterpret_cast<_Float16*>(smem + OFF_H1);
    float*    xs   = reinterpret_cast<float*>(smem + OFF_XS);
    float*    w1s  = reinterpret_cast<float*>(smem + OFF_W1);
    float*    b1s  = reinterpret_cast<float*>(smem + OFF_B1);
    float*    b21s = reinterpret_cast<float*>(smem + OFF_B21);
    float*    b22s = reinterpret_cast<float*>(smem + OFF_B22);
    float*    w31s = reinterpret_cast<float*>(smem + OFF_W31);
    float*    w32s = reinterpret_cast<float*>(smem + OFF_W32);
    float*    b31s = reinterpret_cast<float*>(smem + OFF_B31);
    float*    b32s = reinterpret_cast<float*>(smem + OFF_B32);
    float*    mns  = reinterpret_cast<float*>(smem + OFF_MEAN);
    float*    sds  = reinterpret_cast<float*>(smem + OFF_STD);
    _Float16* h21s = reinterpret_cast<_Float16*>(smem + OFF_H21);
    _Float16* h22s = reinterpret_cast<_Float16*>(smem + OFF_H22);

    const int tid  = threadIdx.x;
    const int row0 = blockIdx.x * ROWS;

    // gfx1250 prefetch hint: next block's x tile (speculative, OOB is dropped)
    __builtin_prefetch(gx + (size_t)(row0 + ROWS) * FN + tid, 0, 1);

    // ---------------- Phase 0: cooperative staging into LDS ----------------
    for (int i = tid; i < ROWS * FN; i += NTHR) {
        int gidx = row0 * FN + i;
        xs[i] = (gidx < Bn * FN) ? gx[gidx] : 0.0f;
    }
    for (int i = tid; i < FN * H1N; i += NTHR) w1s[i] = gW1[i];
    for (int i = tid; i < H1N; i += NTHR)      b1s[i] = gb1[i];
    for (int i = tid; i < H2N; i += NTHR) { b21s[i] = gb21[i]; b22s[i] = gb22[i]; }
    for (int i = tid; i < H2N * 2; i += NTHR) { w31s[i] = gW31[i]; w32s[i] = gW32[i]; }
    if (tid < 2)  { b31s[tid] = gb31[tid]; b32s[tid] = gb32[tid]; }
    if (tid < FN) { mns[tid] = gmean[tid]; sds[tid] = gstd[tid]; }

    // W21/W22 -> f16, stored directly in WMMA B-fragment order:
    // fragment(kk,nt): lane holds 16 contiguous halves; K = kk*32 + (lane>=16?16:0) + h, N = nt*16 + (lane&15)
    for (int i = tid; i < H1N * H2N; i += NTHR) {
        int N  = i & (H2N - 1);
        int K  = i >> 7;                 // row-major [256,128], coalesced global reads
        int kk = K >> 5, r = K & 31;
        int lane = ((r >= 16) ? 16 : 0) | (N & 15);
        int h  = r & 15;
        int nt = N >> 4;
        int L  = (kk * 8 + nt) * 512 + lane * 16 + h;
        w21f[L] = (_Float16)gW21[i];
        w22f[L] = (_Float16)gW22[i];
    }
    __syncthreads();

    // ---------------- Phase 1: h1 = relu(x @ W1 + b1) (K=8, VALU) ----------------
    for (int i = tid; i < ROWS * H1N; i += NTHR) {
        int r = i >> 8;
        int c = i & 255;
        float acc = b1s[c];
#pragma unroll
        for (int k = 0; k < FN; ++k) acc = fmaf(xs[r * FN + k], w1s[k * H1N + c], acc);
        h1s[r * H1S + c] = (_Float16)fmaxf(acc, 0.0f);
    }
    __syncthreads();

    // ---------------- Phase 2: WMMA GEMMs h21/h22 = relu(h1 @ W2x + b2x) ----------------
    {
        const int lane = tid & 31;
        const int hl   = lane >> 4;       // half: 0 = lanes 0-15, 1 = lanes 16-31
        const int l15  = lane & 15;
        const int r0   = (tid >> 5) * 16; // this wave's 16-row M-tile

        // Preload all 8 A-fragments (shared by both weight matrices).
        // 16-bit A 16x32 layout (matches B packing): lanes<16: K0..15; lanes>=16: K16..31
        v16h afr[8];
#pragma unroll
        for (int kk = 0; kk < 8; ++kk) {
            const _Float16* p = h1s + (r0 + l15) * H1S + kk * 32 + hl * 16;
            h8v lo = *reinterpret_cast<const h8v*>(p);
            h8v hi = *reinterpret_cast<const h8v*>(p + 8);
            afr[kk] = __builtin_shufflevector(lo, hi, 0,1,2,3,4,5,6,7,8,9,10,11,12,13,14,15);
        }

        // Two independent WMMA accumulation chains (W21 & W22) per N-tile,
        // with B-fragments double-buffered one K-step ahead so the matrix
        // pipe overlaps ds_load latency instead of draining DScnt to 0.
#pragma unroll
        for (int nt = 0; nt < 8; ++nt) {
            v8f acc21 = {}, acc22 = {};
            v16h b21c = load_bfrag(w21f, 0, nt, lane);
            v16h b22c = load_bfrag(w22f, 0, nt, lane);
#pragma unroll
            for (int kk = 0; kk < 8; ++kk) {
                v16h b21n = b21c, b22n = b22c;
                if (kk < 7) {
                    b21n = load_bfrag(w21f, kk + 1, nt, lane);
                    b22n = load_bfrag(w22f, kk + 1, nt, lane);
                }
                acc21 = __builtin_amdgcn_wmma_f32_16x16x32_f16(
                    false, afr[kk], false, b21c, (short)0, acc21, false, false);
                acc22 = __builtin_amdgcn_wmma_f32_16x16x32_f16(
                    false, afr[kk], false, b22c, (short)0, acc22, false, false);
                b21c = b21n;
                b22c = b22n;
            }
            // C layout: VGPR j -> M = j + 8*half; lane -> N = nt*16 + (lane&15)
            const float bias21 = b21s[nt * 16 + l15];
            const float bias22 = b22s[nt * 16 + l15];
#pragma unroll
            for (int j = 0; j < 8; ++j) {
                float v21 = fmaxf(acc21[j] + bias21, 0.0f);
                float v22 = fmaxf(acc22[j] + bias22, 0.0f);
                int rr = (r0 + j + 8 * hl) * H2S + nt * 16 + l15;
                h21s[rr] = (_Float16)v21;
                h22s[rr] = (_Float16)v22;
            }
        }
    }
    __syncthreads();

    // ---------------- Phase 3: heads + CBF constraints + 25-iter IPM QP ----------------
    if (tid < ROWS && (row0 + tid) < Bn) {
        const int r = tid;
        float p0 = b31s[0], p1 = b31s[1];       // x31 accumulators (QP linear term)
        float q0 = b32s[0], q1 = b32s[1];       // x32 pre-sigmoid
        const _Float16* pr21 = h21s + r * H2S;
        const _Float16* pr22 = h22s + r * H2S;
#pragma unroll 2
        for (int n0 = 0; n0 < H2N; n0 += 8) {
            h8v a21 = *reinterpret_cast<const h8v*>(pr21 + n0);
            h8v a22 = *reinterpret_cast<const h8v*>(pr22 + n0);
#pragma unroll
            for (int q = 0; q < 8; ++q) {
                int n = n0 + q;
                float f21 = (float)a21[q], f22 = (float)a22[q];
                p0 = fmaf(f21, w31s[2 * n], p0);  p1 = fmaf(f21, w31s[2 * n + 1], p1);
                q0 = fmaf(f22, w32s[2 * n], q0);  q1 = fmaf(f22, w32s[2 * n + 1], q1);
            }
        }
        const float pp1 = 4.0f / (1.0f + __expf(-q0));
        const float pp2 = 4.0f / (1.0f + __expf(-q1));

        // x0 = x*std + mean  (features 0..5 used)
        float x0[6];
#pragma unroll
        for (int k = 0; k < 6; ++k) x0[k] = fmaf(xs[r * FN + k], sds[k], mns[k]);
        const float px = x0[0], py = x0[1], th = x0[2], vv = x0[3], ox = x0[4], oy = x0[5];
        const float st = __sinf(th), ct = __cosf(th);

        const float obsx[8] = { 1.0f, -1.5f,  2.5f, -2.0f, 0.5f, 3.0f, -3.0f,  1.5f };
        const float obsy[8] = { 1.0f,  2.0f, -1.0f, -2.0f, 3.0f, 2.0f,  0.5f, -2.5f };
        const float obsr[8] = { 0.3f,  0.4f,  0.5f,  0.3f, 0.4f, 0.35f, 0.45f, 0.3f };

        float G0[9], G1[9], hh[9];
        const float psum = pp1 + pp2, pprod = pp1 * pp2, twov2 = 2.0f * vv * vv;
#pragma unroll
        for (int i = 0; i < 9; ++i) {
            float dx, dy, R;
            if (i < 8) { dx = px - obsx[i]; dy = py - obsy[i]; R = 0.30f + obsr[i]; }
            else       { dx = ox;           dy = oy;           R = 0.55f; }
            float bar  = dx * dx + dy * dy - R * R;
            float bdot = 2.0f * dx * vv * ct + 2.0f * dy * vv * st;
            G0[i] = 2.0f * dx * vv * st - 2.0f * dy * vv * ct;     // -LgLfbu1
            G1[i] = -(2.0f * dx * ct + 2.0f * dy * st);            // -LgLfbu2
            hh[i] = twov2 + psum * bdot + pprod * bar;
        }

        // Primal-dual IPM: min 0.5 z'z + p'z  s.t. G z <= h  (Q = I2)
        float z0 = 0.0f, z1 = 0.0f, s[9], lam[9];
#pragma unroll
        for (int i = 0; i < 9; ++i) { s[i] = 1.0f; lam[i] = 1.0f; }

#pragma unroll 1
        for (int it = 0; it < IPM; ++it) {
            float rd0 = z0 + p0, rd1 = z1 + p1;
            float muS = 0.0f;
#pragma unroll
            for (int i = 0; i < 9; ++i) {
                rd0 = fmaf(G0[i], lam[i], rd0);
                rd1 = fmaf(G1[i], lam[i], rd1);
                muS = fmaf(s[i], lam[i], muS);
            }
            const float sigmu = muS * (0.1f / 9.0f);

            float rp[9], D[9], rcs[9];
            float m00 = 1.0f, m01 = 0.0f, m11 = 1.0f;
            float rhs0 = -rd0, rhs1 = -rd1;
#pragma unroll
            for (int i = 0; i < 9; ++i) {
                rp[i]  = G0[i] * z0 + G1[i] * z1 + s[i] - hh[i];
                D[i]   = lam[i] / s[i];
                rcs[i] = (lam[i] * s[i] - sigmu) / s[i];          // r_c / s
                m00 = fmaf(D[i] * G0[i], G0[i], m00);
                m01 = fmaf(D[i] * G0[i], G1[i], m01);
                m11 = fmaf(D[i] * G1[i], G1[i], m11);
                float t = D[i] * rp[i] - rcs[i];
                rhs0 = fmaf(-G0[i], t, rhs0);
                rhs1 = fmaf(-G1[i], t, rhs1);
            }
            const float inv = 1.0f / (m00 * m11 - m01 * m01);      // M is SPD
            const float dz0 = ( m11 * rhs0 - m01 * rhs1) * inv;
            const float dz1 = (-m01 * rhs0 + m00 * rhs1) * inv;

            float a_s = 1.0e9f, a_l = 1.0e9f;
            float dlam[9], dsv[9];
#pragma unroll
            for (int i = 0; i < 9; ++i) {
                float dl  = D[i] * (G0[i] * dz0 + G1[i] * dz1 + rp[i]) - rcs[i];
                float rc  = lam[i] * s[i] - sigmu;
                float dsi = -(rc + s[i] * dl) / lam[i];
                dlam[i] = dl; dsv[i] = dsi;
                if (dsi < 0.0f) a_s = fminf(a_s, -s[i] / dsi);
                if (dl  < 0.0f) a_l = fminf(a_l, -lam[i] / dl);
            }
            const float alpha = fminf(1.0f, 0.99f * fminf(a_s, a_l));
            z0 = fmaf(alpha, dz0, z0); z1 = fmaf(alpha, dz1, z1);
#pragma unroll
            for (int i = 0; i < 9; ++i) {
                s[i]   = fmaf(alpha, dsv[i],  s[i]);
                lam[i] = fmaf(alpha, dlam[i], lam[i]);
            }
        }
        out[(size_t)(row0 + r) * 2 + 0] = z0;
        out[(size_t)(row0 + r) * 2 + 1] = z1;
    }
}

extern "C" void kernel_launch(void* const* d_in, const int* in_sizes, int n_in,
                              void* d_out, int out_size, void* d_ws, size_t ws_size,
                              hipStream_t stream) {
    (void)n_in; (void)out_size; (void)d_ws; (void)ws_size;
    const float* x    = (const float*)d_in[0];
    const float* mean = (const float*)d_in[1];
    const float* stdv = (const float*)d_in[2];
    const float* W1   = (const float*)d_in[3];
    const float* b1   = (const float*)d_in[4];
    const float* W21  = (const float*)d_in[5];
    const float* b21  = (const float*)d_in[6];
    const float* W22  = (const float*)d_in[7];
    const float* b22  = (const float*)d_in[8];
    const float* W31  = (const float*)d_in[9];
    const float* b31  = (const float*)d_in[10];
    const float* W32  = (const float*)d_in[11];
    const float* b32  = (const float*)d_in[12];
    float* out = (float*)d_out;

    const int Bn   = in_sizes[0] / FN;
    const int grid = (Bn + ROWS - 1) / ROWS;

    // ~278 KB dynamic LDS (WGP has 320 KB): raise the per-kernel cap.
    (void)hipFuncSetAttribute((const void*)barriernet_fused,
                              hipFuncAttributeMaxDynamicSharedMemorySize, SMEMB);

    hipLaunchKernelGGL(barriernet_fused, dim3(grid), dim3(NTHR), SMEMB, stream,
                       x, mean, stdv, W1, b1, W21, b21, W22, b22,
                       W31, b31, W32, b32, out, Bn);
}
